// SlimEmbedding_43671227466010
// MI455X (gfx1250) — matrix operand order repaired and checked by reading, not compiled, not run on hardware
//
#include <hip/hip_runtime.h>
#include <hip/hip_bf16.h>

// ---- problem constants (match reference) ----
#define V_DIM 100000
#define R_DIM 256
#define D_DIM 1024
#define A_STRIDE 264   // Atile row stride in bf16: 528B -> lane L hits bank
                       // group (4L)%64, killing the 16-way conflict that a
                       // 512B stride causes on 64x4B LDS banks.

typedef __attribute__((ext_vector_type(8)))  __bf16 v8bf;
typedef __attribute__((ext_vector_type(16))) __bf16 v16bf;
typedef __attribute__((ext_vector_type(4)))  float  v4f;
typedef __attribute__((ext_vector_type(8)))  float  v8f;

// ---------------------------------------------------------------------------
// Kernel 1: Wt[n][k] = bf16( s[k] * Vt[k][n] )   (scale folded into B, plus
// transpose so B-matrix fragments are K-contiguous -> b128 loads).
// LDS-tiled 32x32 transpose, coalesced on both sides.
// ---------------------------------------------------------------------------
__global__ void __launch_bounds__(256)
SlimEmbedding_prep_wt(const float* __restrict__ Vt, const float* __restrict__ s,
                      __bf16* __restrict__ Wt) {
  __shared__ float tile[32][33];          // +1 pad kills bank conflicts
  const int tx = threadIdx.x;             // 0..31
  const int ty = threadIdx.y;             // 0..7
  const int n0 = blockIdx.x * 32;
  const int k0 = blockIdx.y * 32;
#pragma unroll
  for (int j = 0; j < 4; ++j) {
    const int k = k0 + ty + j * 8;
    tile[ty + j * 8][tx] = s[k] * Vt[(size_t)k * D_DIM + n0 + tx];
  }
  __syncthreads();
#pragma unroll
  for (int j = 0; j < 4; ++j) {
    const int n = n0 + ty + j * 8;
    Wt[(size_t)n * R_DIM + k0 + tx] = (__bf16)tile[tx][ty + j * 8];
  }
}

// ---------------------------------------------------------------------------
// Kernel 2: gathered GEMM,  out[tok,:] = bf16(U[ids[tok],:]) @ Wt^T  (f32 acc)
// Block = 256 threads = 8 waves; covers 64 tokens (4 M-tiles) x 256 columns.
// Each wave: 4M x 2N register blocking -> each B fragment feeds 4 WMMAs.
// Per K-step: preload 4 A fragments (one ds clause), then 8 back-to-back WMMAs.
// ---------------------------------------------------------------------------
__global__ void __launch_bounds__(256)
SlimEmbedding_gemm(const int* __restrict__ ids, const float* __restrict__ U,
                   const __bf16* __restrict__ Wt, float* __restrict__ out) {
  __shared__ __align__(16) __bf16 Atile[64 * A_STRIDE];  // ~33 KB, padded rows

  const int t       = threadIdx.x;
  const int mBlock  = blockIdx.x >> 2;          // 64-token block index
  const int colSlab = (blockIdx.x & 3) * 256;   // 256-column slab

  // ---- gather 64 rows of U, convert fp32 -> bf16, stage in LDS ----
#pragma unroll
  for (int rr = 0; rr < 4; ++rr) {
    const int task = t + rr * 256;     // 1024 tasks: 64 rows x 16 segments
    const int row  = task >> 4;        // 0..63
    const int seg  = (task & 15) * 16; // 16-element K segment
    const int tok  = ids[mBlock * 64 + row];
    const float* up = U + (size_t)tok * R_DIM + seg;
    v4f f0 = *(const v4f*)(up + 0);
    v4f f1 = *(const v4f*)(up + 4);
    v4f f2 = *(const v4f*)(up + 8);
    v4f f3 = *(const v4f*)(up + 12);
    v8bf b0, b1;
#pragma unroll
    for (int i = 0; i < 4; ++i) {
      b0[i]     = (__bf16)f0[i];
      b0[i + 4] = (__bf16)f1[i];
      b1[i]     = (__bf16)f2[i];
      b1[i + 4] = (__bf16)f3[i];
    }
    *(v8bf*)&Atile[row * A_STRIDE + seg]     = b0;
    *(v8bf*)&Atile[row * A_STRIDE + seg + 8] = b1;
  }
  __syncthreads();

  const int lane = t & 31;
  const int wave = t >> 5;     // 0..7
  const int lo16 = lane & 15;
  const int hi   = lane >> 4;  // 0 or 1 (lane half)

  const int n0 = colSlab + wave * 32;  // this wave's first N tile
  const int n1 = n0 + 16;              // second N tile

  v8f c[4][2] = {};                    // 4M x 2N accumulators (64 VGPRs)

  // A k order used here (not ISA table order): lane half -> K 0..15 / 16..31, elements
  // K-consecutive (matches B fragment order).
  const __bf16* aBase  = &Atile[lo16 * A_STRIDE + hi * 16];
  // B k order used here (same as A): lane = column; lane half -> K 0..15 / 16..31,
  // elements K-consecutive -> one 32B run in Wt[n][k] (two b128 loads).
  const __bf16* b0Base = Wt + (size_t)(n0 + lo16) * R_DIM + hi * 16;
  const __bf16* b1Base = Wt + (size_t)(n1 + lo16) * R_DIM + hi * 16;

#pragma unroll
  for (int k0 = 0; k0 < R_DIM; k0 += 32) {
    v16bf b0 = *(const v16bf*)(b0Base + k0);
    v16bf b1 = *(const v16bf*)(b1Base + k0);

    // Preload all 4 A fragments for this K-step (batched ds clause).
    v16bf a[4];
#pragma unroll
    for (int m = 0; m < 4; ++m) {
      const __bf16* ap = aBase + m * 16 * A_STRIDE + k0;
      v8bf alo = *(const v8bf*)(ap);
      v8bf ahi = *(const v8bf*)(ap + 8);
      a[m] = __builtin_shufflevector(alo, ahi,
          0, 1, 2, 3, 4, 5, 6, 7, 8, 9, 10, 11, 12, 13, 14, 15);
    }

    // 8 back-to-back WMMAs; (neg_a, A, neg_b, B, c_mod, C, reuse_a, reuse_b)
#pragma unroll
    for (int m = 0; m < 4; ++m) {
      c[m][0] = __builtin_amdgcn_wmma_f32_16x16x32_bf16(false, a[m], false, b0,
                                                (short)0, c[m][0], false, false);
      c[m][1] = __builtin_amdgcn_wmma_f32_16x16x32_bf16(false, a[m], false, b1,
                                                (short)0, c[m][1], false, false);
    }
  }

  // C/D f32 16x16 layout: VGPR v -> row v (lanes 0-15) / v+8 (lanes 16-31),
  // column = lane & 15.
  float* obase = out + ((size_t)mBlock * 64 + hi * 8) * D_DIM + lo16;
#pragma unroll
  for (int m = 0; m < 4; ++m) {
#pragma unroll
    for (int v = 0; v < 8; ++v) {
      obase[(size_t)(m * 16 + v) * D_DIM + n0] = c[m][0][v];
      obase[(size_t)(m * 16 + v) * D_DIM + n1] = c[m][1][v];
    }
  }
}

// ---------------------------------------------------------------------------
extern "C" void kernel_launch(void* const* d_in, const int* in_sizes, int n_in,
                              void* d_out, int out_size, void* d_ws, size_t ws_size,
                              hipStream_t stream) {
  const int*   ids = (const int*)d_in[0];    // [B,S] int32 token ids
  const float* U   = (const float*)d_in[1];  // [V,R] fp32
  const float* s   = (const float*)d_in[2];  // [R]   fp32
  const float* Vt  = (const float*)d_in[3];  // [R,D] fp32
  float*       out = (float*)d_out;          // [B,S,D] fp32
  __bf16*      Wt  = (__bf16*)d_ws;          // [D,R] bf16 scratch (512 KB)

  // 1) fold s into Vt, transpose, downconvert to bf16
  dim3 pb(32, 8);
  dim3 pg(D_DIM / 32, R_DIM / 32);
  SlimEmbedding_prep_wt<<<pg, pb, 0, stream>>>(Vt, s, Wt);

  // 2) gathered GEMM via bf16 WMMA, 4Mx2N register blocking per wave
  const int tokens  = in_sizes[0];           // B*S = 16384
  const int mBlocks = tokens / 64;           // 256
  dim3 gb(256);
  dim3 gg(mBlocks * (D_DIM / 256));          // 1024 blocks
  SlimEmbedding_gemm<<<gg, gb, 0, stream>>>(ids, U, Wt, out);
}
